// SRUModel_704374637371
// MI455X (gfx1250) — matrix operand order repaired and checked by reading, not compiled, not run on hardware
//
#include <hip/hip_runtime.h>
#include <hip/hip_bf16.h>
#include <stdint.h>

// ---------------------------------------------------------------------------
// SRU language model forward for MI455X (gfx1250, wave32, WMMA + async LDS).
//   x = emb[inputs]                      (rows indexed row = b*L + l)
//   layer0: U0 = x @ W0   (k=4)  -> recurrence -> h0
//   layer1: U1 = h0 @ W1  (k=3, skip = h0) -> recurrence -> h1
//   out    = h1 @ Wp + bp                (134 GFLOP: dominant, WMMA bf16)
// Weights are pre-transposed to [N][K] bf16 so GEMM staging is pure b128
// global->LDS async DMA (ASYNCcnt), double buffered.
// ---------------------------------------------------------------------------

#define VOCAB 32000
#define EMB   64
#define HID   512
#define BB    8
#define LL    512
#define MROWS (BB * LL)   // 4096

typedef __bf16  bf16_t;
typedef __attribute__((ext_vector_type(16))) __bf16 v16bf;
typedef __attribute__((ext_vector_type(8)))  __bf16 v8bf;
typedef __attribute__((ext_vector_type(8)))  float  v8f;

// ---- gfx1250 async global->LDS copy (ASYNCcnt), with safe fallback --------
#if __has_builtin(__builtin_amdgcn_global_load_async_to_lds_b128)
#define HAVE_ASYNC_LDS 1
#else
#define HAVE_ASYNC_LDS 0
#endif

// Builtin signature (from hipcc diagnostic): param0 = v4i addrspace(1)*,
// where v4i is gcc-style 'int __attribute__((vector_size(16)))'.
typedef int v4i_t __attribute__((vector_size(16)));
typedef __attribute__((address_space(1))) v4i_t* g_v4i_ptr;
typedef __attribute__((address_space(3))) v4i_t* l_v4i_ptr;

__device__ __forceinline__ void copy16_g2l(const bf16_t* g, bf16_t* l) {
#if HAVE_ASYNC_LDS
  __builtin_amdgcn_global_load_async_to_lds_b128((g_v4i_ptr)(void*)g,
                                                 (l_v4i_ptr)(void*)l, 0, 0);
#else
  *(v8bf*)l = *(const v8bf*)g;
#endif
}

__device__ __forceinline__ void wait_async_copies() {
#if HAVE_ASYNC_LDS
#if __has_builtin(__builtin_amdgcn_s_wait_asynccnt)
  __builtin_amdgcn_s_wait_asynccnt(0);
#else
  asm volatile("s_wait_asynccnt 0" ::: "memory");
#endif
#endif
}

// ---------------------------------------------------------------------------
// Embedding gather + f32 -> bf16 cast.  rows are b*L + l, inputs is (B,L)
// row-major, so token index == row index.
// ---------------------------------------------------------------------------
__global__ __launch_bounds__(256) void gather_cast_kernel(
    const int* __restrict__ inp, const float* __restrict__ emb,
    bf16_t* __restrict__ xb) {
  int idx = blockIdx.x * 256 + threadIdx.x;       // element id in (4096 x 64)
  if (idx >= MROWS * EMB) return;
  int row = idx / EMB;
  int e   = idx - row * EMB;
  int tok = inp[row];
  xb[idx] = (bf16_t)emb[(size_t)tok * EMB + e];
}

// ---------------------------------------------------------------------------
// Weight transpose + cast: src f32 [K][N] -> dst bf16 [N][K].
// LDS-tiled 32x32 so both global sides stay coalesced. K,N % 32 == 0.
// ---------------------------------------------------------------------------
__global__ __launch_bounds__(256) void transpose_cast_kernel(
    const float* __restrict__ src, bf16_t* __restrict__ dst, int K, int N) {
  __shared__ float tile[32][33];
  const int tx = threadIdx.x & 31;
  const int ty = threadIdx.x >> 5;               // 0..7
  const int k0 = blockIdx.y * 32;
  const int n0 = blockIdx.x * 32;
#pragma unroll
  for (int i = 0; i < 32; i += 8)
    tile[ty + i][tx] = src[(size_t)(k0 + ty + i) * N + n0 + tx];
  __syncthreads();
#pragma unroll
  for (int i = 0; i < 32; i += 8)
    dst[(size_t)(n0 + ty + i) * K + k0 + tx] = (bf16_t)tile[tx][ty + i];
}

// ---------------------------------------------------------------------------
// Tiled bf16 GEMM with f32 accumulate via v_wmma_f32_16x16x32_bf16.
//   C[M,N] = A[M,K] * Bt[N,K]^T (+ bias[N] if bias != nullptr)
// Block: 256 threads = 8 waves. Block tile 128(M) x 128(N), K step 32.
// Wave grid 2(M) x 4(N); each wave computes 64x32 = 4x2 WMMA tiles
// (8 v_wmma per K step, 64 accumulation VGPRs per lane).
// Double-buffered LDS (32 KB), staged entirely with async b128 copies.
// Requires M%128==0, N%128==0, K%32==0 (true for all three GEMMs here).
// ---------------------------------------------------------------------------
#define BM 128
#define BN 128
#define BK 32

__global__ __launch_bounds__(256) void gemm_bf16_wmma_kernel(
    const bf16_t* __restrict__ A,   // [M][K]
    const bf16_t* __restrict__ Bt,  // [N][K] (pre-transposed weights)
    float* __restrict__ C, const float* __restrict__ bias,
    int M, int N, int K) {
  __shared__ __align__(16) bf16_t sA[2][BM][BK];   // 2 x 8 KB
  __shared__ __align__(16) bf16_t sB[2][BN][BK];   // 2 x 8 KB

  const int tid   = threadIdx.x;
  const int lane  = tid & 31;
  const int wave  = tid >> 5;      // 0..7
  const int waveM = wave & 1;      // 0..1 -> m offset waveM*64
  const int waveN = wave >> 1;     // 0..3 -> n offset waveN*32
  const int row16 = lane & 15;
  const int half  = lane >> 4;     // 0 or 1

  const int mBase = blockIdx.y * BM;
  const int nBase = blockIdx.x * BN;

  // Per-thread staging coordinates (all b128 chunks; 2 for A, 2 for B).
  const int aR0 = tid >> 2;                 // rows 0..63 (chunk 1: +64)
  const int aK  = (tid & 3) << 3;
  const int bR0 = tid >> 2;                 // n rows 0..63 (chunk 1: +64)
  const int bK  = (tid & 3) << 3;

  auto stage = [&](int buf, int k0) {
    copy16_g2l(&A[(size_t)(mBase + aR0) * K + k0 + aK], &sA[buf][aR0][aK]);
    copy16_g2l(&A[(size_t)(mBase + aR0 + 64) * K + k0 + aK],
               &sA[buf][aR0 + 64][aK]);
    copy16_g2l(&Bt[(size_t)(nBase + bR0) * K + k0 + bK], &sB[buf][bR0][bK]);
    copy16_g2l(&Bt[(size_t)(nBase + bR0 + 64) * K + k0 + bK],
               &sB[buf][bR0 + 64][bK]);
  };

  v8f acc[4][2] = {};

  stage(0, 0);
  wait_async_copies();
  __syncthreads();

  int buf = 0;
  for (int k0 = 0; k0 < K; k0 += BK) {
    const int nk = k0 + BK;
    if (nk < K) {
      stage(buf ^ 1, nk);                       // overlap DMA with math
      if (nk + BK < K) {                        // L2 prefetch two tiles ahead
        __builtin_prefetch(&A[(size_t)(mBase + aR0) * K + nk + BK], 0, 3);
        __builtin_prefetch(&Bt[(size_t)(nBase + bR0) * K + nk + BK], 0, 3);
      }
    }

    // ---- fragments (ISA 7.12.2 bf16 A/B layouts, wave32) ------------------
    v16bf aFrag[4], bFrag[2];
#pragma unroll
    for (int tm = 0; tm < 4; ++tm) {
      int r = waveM * 64 + tm * 16 + row16;
      v8bf lo = *(const v8bf*)&sA[buf][r][half * 8];      // K=half*8+0..7
      v8bf hi = *(const v8bf*)&sA[buf][r][16 + half * 8]; // K=16+half*8+0..7
      aFrag[tm] = __builtin_shufflevector(lo, hi, 0, 1, 2, 3, 4, 5, 6, 7, 8,
                                          9, 10, 11, 12, 13, 14, 15);
    }
#pragma unroll
    for (int tn = 0; tn < 2; ++tn) {
      int nc = waveN * 32 + tn * 16 + row16;
      v8bf lo = *(const v8bf*)&sB[buf][nc][half * 8];      // K=half*8+0..7
      v8bf hi = *(const v8bf*)&sB[buf][nc][16 + half * 8]; // K=16+half*8+0..7
      bFrag[tn] = __builtin_shufflevector(lo, hi, 0, 1, 2, 3, 4, 5, 6, 7, 8,
                                          9, 10, 11, 12, 13, 14, 15);
    }

#pragma unroll
    for (int tm = 0; tm < 4; ++tm)
#pragma unroll
      for (int tn = 0; tn < 2; ++tn)
        acc[tm][tn] = __builtin_amdgcn_wmma_f32_16x16x32_bf16(
            false, aFrag[tm], false, bFrag[tn], (short)0, acc[tm][tn],
            false, false);

    if (nk < K) {
      wait_async_copies();    // next-tile DMA landed in LDS
      __syncthreads();        // visible to all waves
      buf ^= 1;
    }
  }

  // ---- store C (+bias). Lane l holds N = l%16, M = vgpr + 8*(l/16). -------
#pragma unroll
  for (int tn = 0; tn < 2; ++tn) {
    int n = nBase + waveN * 32 + tn * 16 + row16;
    float bv = bias ? bias[n] : 0.0f;
#pragma unroll
    for (int tm = 0; tm < 4; ++tm) {
      int m0 = mBase + waveM * 64 + tm * 16 + half * 8;
#pragma unroll
      for (int v = 0; v < 8; ++v)
        C[(size_t)(m0 + v) * N + n] = acc[tm][tn][v] + bv;
    }
  }
}

// ---------------------------------------------------------------------------
// SRU recurrence. One thread per (b, h); sequential over L.
//   c_t = f*c + (1-f)*x~ ;  out = r*tanh(c) + (1-r)*skip
// kGroups==4: skip = U[...,3];  kGroups==3: skip = skipSrc (layer input h0).
// ---------------------------------------------------------------------------
__global__ __launch_bounds__(256) void sru_kernel(
    const float* __restrict__ U, const float* __restrict__ bvec,
    const float* __restrict__ skipSrc, float* __restrict__ outF,
    bf16_t* __restrict__ outB, int kGroups) {
  int idx  = blockIdx.x * 256 + threadIdx.x;   // 0..4095
  int b    = idx >> 9;                         // batch
  int h    = idx & (HID - 1);
  float bf = bvec[h];
  float br = bvec[HID + h];
  float c  = 0.0f;
  int ustride = kGroups * HID;
  for (int l = 0; l < LL; ++l) {
    size_t row = (size_t)b * LL + l;
    const float* u = U + row * ustride;
    float xt = u[h];
    float f  = 1.0f / (1.0f + __expf(-(u[HID + h] + bf)));
    float r  = 1.0f / (1.0f + __expf(-(u[2 * HID + h] + br)));
    float sk = (kGroups == 4) ? u[3 * HID + h] : skipSrc[row * HID + h];
    c = f * c + (1.0f - f) * xt;
    float o = r * tanhf(c) + (1.0f - r) * sk;
    if (outF) outF[row * HID + h] = o;
    outB[row * HID + h] = (bf16_t)o;
  }
}

// ---------------------------------------------------------------------------
// Host-side launch.
// ---------------------------------------------------------------------------
extern "C" void kernel_launch(void* const* d_in, const int* in_sizes, int n_in,
                              void* d_out, int out_size, void* d_ws,
                              size_t ws_size, hipStream_t stream) {
  const int*   inp = (const int*)d_in[0];
  const float* emb = (const float*)d_in[1];
  const float* W0  = (const float*)d_in[2];
  const float* b0  = (const float*)d_in[3];
  const float* W1  = (const float*)d_in[4];
  const float* b1  = (const float*)d_in[5];
  const float* Wp  = (const float*)d_in[6];
  const float* bp  = (const float*)d_in[7];
  float* out = (float*)d_out;

  char*  ws  = (char*)d_ws;
  size_t off = 0;
  auto alloc = [&](size_t bytes) -> void* {
    void* p = ws + off;
    off = (off + bytes + 255) & ~(size_t)255;
    return p;
  };

  bf16_t* xb  = (bf16_t*)alloc((size_t)MROWS * EMB * 2);        // 512 KB
  bf16_t* W0t = (bf16_t*)alloc((size_t)EMB * 4 * HID * 2);      // 256 KB  [2048][64]
  bf16_t* W1t = (bf16_t*)alloc((size_t)HID * 3 * HID * 2);      // 1.5 MB  [1536][512]
  bf16_t* Wpt = (bf16_t*)alloc((size_t)HID * VOCAB * 2);        // 31.25 MB [32000][512]
  float*  h0f = (float*)alloc((size_t)MROWS * HID * 4);         // 8 MB
  bf16_t* h0b = (bf16_t*)alloc((size_t)MROWS * HID * 2);        // 4 MB
  bf16_t* h1b = (bf16_t*)alloc((size_t)MROWS * HID * 2);        // 4 MB
  float*  Ubuf = (float*)alloc((size_t)MROWS * 4 * HID * 4);    // 32 MB (U0, reused for U1)

  // 1) embedding gather + cast
  gather_cast_kernel<<<(MROWS * EMB + 255) / 256, 256, 0, stream>>>(inp, emb, xb);

  // 2) weight transpose+cast to bf16 [N][K]
  transpose_cast_kernel<<<dim3((4 * HID) / 32, EMB / 32), 256, 0, stream>>>(
      W0, W0t, EMB, 4 * HID);
  transpose_cast_kernel<<<dim3((3 * HID) / 32, HID / 32), 256, 0, stream>>>(
      W1, W1t, HID, 3 * HID);
  transpose_cast_kernel<<<dim3(VOCAB / 32, HID / 32), 256, 0, stream>>>(
      Wp, Wpt, HID, VOCAB);

  // 3) U0 = x @ W0   (4096 x 2048, K=64)
  {
    dim3 grid((4 * HID) / BN, MROWS / BM);
    gemm_bf16_wmma_kernel<<<grid, 256, 0, stream>>>(xb, W0t, Ubuf, nullptr,
                                                    MROWS, 4 * HID, EMB);
  }

  // 4) SRU layer 0 (k=4) -> h0 (f32 + bf16)
  sru_kernel<<<MROWS / 256, 256, 0, stream>>>(Ubuf, b0, nullptr, h0f, h0b, 4);

  // 5) U1 = h0 @ W1  (4096 x 1536, K=512)   [reuses Ubuf]
  {
    dim3 grid((3 * HID) / BN, MROWS / BM);
    gemm_bf16_wmma_kernel<<<grid, 256, 0, stream>>>(h0b, W1t, Ubuf, nullptr,
                                                    MROWS, 3 * HID, HID);
  }

  // 6) SRU layer 1 (k=3, skip = h0) -> h1 (bf16 only)
  sru_kernel<<<MROWS / 256, 256, 0, stream>>>(Ubuf, b1, h0f, nullptr, h1b, 3);

  // 7) out = h1 @ Wp + bp  (4096 x 32000, K=512) — dominant GEMM
  {
    dim3 grid(VOCAB / BN, MROWS / BM);
    gemm_bf16_wmma_kernel<<<grid, 256, 0, stream>>>(h1b, Wpt, out, bp,
                                                    MROWS, VOCAB, HID);
  }
}
